// TreeLSTM_67963562492255
// MI455X (gfx1250) — matrix operand order repaired and checked by reading, compiled, not run on hardware
//
#include <hip/hip_runtime.h>

#define LL  20
#define NN  1024
#define INW 64
#define HH  512
#define G4  2048          // 4*H
#define F2P 1152          // feat width 1147 padded to /32
#define M1  (LL * NN)     // 20480
#define B1  (2 * M1)      // 40960 (cond1 + cond2 batches)

typedef __attribute__((ext_vector_type(16))) __bf16 v16bf;
typedef __attribute__((ext_vector_type(8)))  __bf16 v8bf;
typedef __attribute__((ext_vector_type(8)))  float  v8f;

__device__ __forceinline__ v8f vzero8() {
  v8f z;
#pragma unroll
  for (int i = 0; i < 8; ++i) z[i] = 0.f;
  return z;
}

__device__ __forceinline__ v8f wmma_bf16(v16bf a, v16bf b, v8f c) {
  return __builtin_amdgcn_wmma_f32_16x16x32_bf16(false, a, false, b, (short)0, c,
                                                 false, false);
}

// A fragment (16x32 bf16) from an LDS tile A[16][lda], row-major.
// lane<16: M=lane, K=kb..kb+15; lane>=16: M=lane-16, K=kb+16..kb+31.
__device__ __forceinline__ v16bf loadA(const __bf16* A, int lda, int kbase, int lane) {
  const int m = lane & 15;
  const int koff = (lane < 16) ? 0 : 16;
  const __bf16* p = A + m * lda + kbase + koff;
  v8bf lo = *(const v8bf*)p;
  v8bf hi = *(const v8bf*)(p + 8);
  return __builtin_shufflevector(lo, hi, 0, 1, 2, 3, 4, 5, 6, 7,
                                 8, 9, 10, 11, 12, 13, 14, 15);
}

// B fragment (32x16) for D = A @ W^T: W[Nout][ldw] row-major (K contiguous).
// lane<16: N=lane, K=kb..kb+15; lane>=16: N=lane-16, K=kb+16..kb+31.  32B/lane.
__device__ __forceinline__ v16bf loadB(const __bf16* W, int ldw, int n0, int kbase,
                                       int lane) {
  const int n = lane & 15;
  const int koff = (lane < 16) ? 0 : 16;
  return *(const v16bf*)(W + (size_t)(n0 + n) * ldw + kbase + koff);
}

__device__ __forceinline__ float sigmf(float x) { return 1.f / (1.f + __expf(-x)); }

// ---------------- weight prep ----------------
__global__ void k_cvt_pad(const float* __restrict__ src, __bf16* __restrict__ dst,
                          int rows, int K, int Kpad) {
  const long long tot = (long long)rows * Kpad;
  const long long stride = (long long)gridDim.x * blockDim.x;
  for (long long i = blockIdx.x * (long long)blockDim.x + threadIdx.x; i < tot; i += stride) {
    long long r = i / Kpad;
    int k = (int)(i - r * Kpad);
    dst[i] = (k < K) ? (__bf16)src[r * K + k] : (__bf16)0.f;
  }
}

__global__ void k_bias_sum(const float* __restrict__ a, const float* __restrict__ b,
                           float* __restrict__ o, int n) {
  int i = blockIdx.x * blockDim.x + threadIdx.x;
  if (i < n) o[i] = a[i] + b[i];
}

// ---------------- LSTM1: 40960 seqs x 8 steps, IN=64, H=512 ----------------
__global__ void __launch_bounds__(256)
k_lstm1(const float* __restrict__ c1, const float* __restrict__ c2,
        const __bf16* __restrict__ Wih, const __bf16* __restrict__ Whh,
        const float* __restrict__ bsum, __bf16* __restrict__ hout) {
  extern __shared__ char smem[];
  float*  C  = (float*)smem;                 // [32][512] f32
  __bf16* Hb = (__bf16*)(smem + 32 * 512 * 4);  // [2][32][520] bf16 (ping-pong)
  __bf16* X  = Hb + 2 * 32 * 520;               // [8][32][72] bf16
  const int tid = threadIdx.x, lane = tid & 31, wave = tid >> 5;
  const size_t r0 = (size_t)blockIdx.x * 32;

  for (int i = tid; i < 32 * 512; i += 256) {
    int row = i >> 9, kk = i & 511;
    size_t r = r0 + row;
    float v = (r < (size_t)M1) ? c1[r * 512 + kk] : c2[(r - M1) * 512 + kk];
    int t = kk >> 6, k = kk & 63;
    X[(t * 32 + row) * 72 + k] = (__bf16)v;
    C[i] = 0.f;
  }
  for (int i = tid; i < 2 * 32 * 520; i += 256) Hb[i] = (__bf16)0.f;
  __syncthreads();

  const int mo = (lane < 16) ? 0 : 8;
  const int nn = lane & 15;

  for (int t = 0; t < 8; ++t) {
    const __bf16* Hr = Hb + (t & 1) * (32 * 520);
    __bf16*       Hw = Hb + ((t + 1) & 1) * (32 * 520);
    const __bf16* Xt = X + t * (32 * 72);
    for (int pi = 0; pi < 8; ++pi) {
      const int p = wave * 8 + pi;
      const int mb = p >> 5, cg = p & 31;
      v8f acc[4];
#pragma unroll
      for (int g = 0; g < 4; ++g) acc[g] = vzero8();
#pragma unroll
      for (int ks = 0; ks < 2; ++ks) {                  // x @ Wih^T  (K=64)
        v16bf a = loadA(Xt + mb * 16 * 72, 72, ks * 32, lane);
#pragma unroll
        for (int g = 0; g < 4; ++g) {
          v16bf b = loadB(Wih, INW, g * HH + cg * 16, ks * 32, lane);
          acc[g] = wmma_bf16(a, b, acc[g]);
        }
      }
#pragma unroll 4
      for (int ks = 0; ks < 16; ++ks) {                 // h @ Whh^T  (K=512)
        v16bf a = loadA(Hr + mb * 16 * 520, 520, ks * 32, lane);
#pragma unroll
        for (int g = 0; g < 4; ++g) {
          v16bf b = loadB(Whh, HH, g * HH + cg * 16, ks * 32, lane);
          acc[g] = wmma_bf16(a, b, acc[g]);
        }
      }
      const int col = cg * 16 + nn;
      const float bi = bsum[col], bff = bsum[HH + col];
      const float bgg = bsum[2 * HH + col], boo = bsum[3 * HH + col];
#pragma unroll
      for (int rr = 0; rr < 8; ++rr) {
        const int row = mb * 16 + mo + rr;
        float zi = acc[0][rr] + bi, zf = acc[1][rr] + bff;
        float zg = acc[2][rr] + bgg, zo = acc[3][rr] + boo;
        float cv = C[row * 512 + col];
        cv = sigmf(zf) * cv + sigmf(zi) * tanhf(zg);
        C[row * 512 + col] = cv;
        Hw[row * 520 + col] = (__bf16)(sigmf(zo) * tanhf(cv));
      }
    }
    __syncthreads();
  }
  const __bf16* Hf = Hb;  // after t=7 final h sits in buffer 0
  for (int i = tid; i < 32 * 512; i += 256) {
    int row = i >> 9, k = i & 511;
    hout[(r0 + row) * 512 + k] = Hf[row * 520 + k];
  }
}

// ---------------- lo = (relu(h1 Wc^T+bc)+relu(h2 Wc^T+bc))/2 ----------------
__global__ void __launch_bounds__(256)
k_logemm(const __bf16* __restrict__ h1, const __bf16* __restrict__ Wc,
         const float* __restrict__ bc, float* __restrict__ lo) {
  extern __shared__ char smem[];
  __bf16* A1 = (__bf16*)smem;      // [32][520]
  __bf16* A2 = A1 + 32 * 520;      // [32][520]
  const int tid = threadIdx.x, lane = tid & 31, wave = tid >> 5;
  const size_t r0 = (size_t)blockIdx.x * 32;
  for (int i = tid; i < 32 * 512; i += 256) {
    int row = i >> 9, k = i & 511;
    A1[row * 520 + k] = h1[(r0 + row) * 512 + k];
    A2[row * 520 + k] = h1[((size_t)M1 + r0 + row) * 512 + k];
  }
  __syncthreads();
  const int mo = (lane < 16) ? 0 : 8, nn = lane & 15;
  for (int pi = 0; pi < 8; ++pi) {
    const int p = wave * 8 + pi, mb = p >> 5, cg = p & 31;
    v8f acc1 = vzero8(), acc2 = vzero8();
#pragma unroll 4
    for (int ks = 0; ks < 16; ++ks) {
      v16bf b = loadB(Wc, HH, cg * 16, ks * 32, lane);
      v16bf a1 = loadA(A1 + mb * 16 * 520, 520, ks * 32, lane);
      v16bf a2 = loadA(A2 + mb * 16 * 520, 520, ks * 32, lane);
      acc1 = wmma_bf16(a1, b, acc1);
      acc2 = wmma_bf16(a2, b, acc2);
    }
    const int col = cg * 16 + nn;
    const float bb = bc[col];
#pragma unroll
    for (int rr = 0; rr < 8; ++rr) {
      const int row = mb * 16 + mo + rr;
      float v1 = fmaxf(acc1[rr] + bb, 0.f);
      float v2 = fmaxf(acc2[rr] + bb, 0.f);
      lo[(r0 + row) * 512 + col] = 0.5f * (v1 + v2);
    }
  }
}

// ---------------- column mean / rsqrt(var+eps) over M rows ----------------
__global__ void k_colstats(const float* __restrict__ x, int M, int Ncols,
                           float* __restrict__ mean, float* __restrict__ rstd) {
  const int j = blockIdx.x;
  float s = 0.f, sq = 0.f;
  for (int i = threadIdx.x; i < M; i += blockDim.x) {
    float v = x[(size_t)i * Ncols + j];
    s += v;
    sq += v * v;
  }
  __shared__ float bs[256], bq[256];
  bs[threadIdx.x] = s; bq[threadIdx.x] = sq;
  __syncthreads();
  for (int o = 128; o > 0; o >>= 1) {
    if ((int)threadIdx.x < o) { bs[threadIdx.x] += bs[threadIdx.x + o]; bq[threadIdx.x] += bq[threadIdx.x + o]; }
    __syncthreads();
  }
  if (threadIdx.x == 0) {
    float m = bs[0] / M;
    float v = bq[0] / M - m * m;
    mean[j] = m;
    rstd[j] = rsqrtf(v + 1e-5f);
  }
}

// bn1-normalize lo into feat columns [123,635)
__global__ void k_bn_feat(const float* __restrict__ x, const float* __restrict__ mean,
                          const float* __restrict__ rstd, const float* __restrict__ g,
                          const float* __restrict__ b, __bf16* __restrict__ feat) {
  const size_t stride = (size_t)gridDim.x * blockDim.x;
  for (size_t e = blockIdx.x * (size_t)blockDim.x + threadIdx.x; e < (size_t)M1 * HH; e += stride) {
    size_t r = e >> 9; int j = (int)(e & 511);
    float v = g[j] * (x[e] - mean[j]) * rstd[j] + b[j];
    feat[r * F2P + 123 + j] = (__bf16)v;
  }
}

// bn-normalize f32 -> bf16 (flat dst)
__global__ void k_bn_bf(const float* __restrict__ x, const float* __restrict__ mean,
                        const float* __restrict__ rstd, const float* __restrict__ g,
                        const float* __restrict__ b, __bf16* __restrict__ dst, int M) {
  const size_t tot = (size_t)M * HH;
  const size_t stride = (size_t)gridDim.x * blockDim.x;
  for (size_t e = blockIdx.x * (size_t)blockDim.x + threadIdx.x; e < tot; e += stride) {
    int j = (int)(e & 511);
    dst[e] = (__bf16)(g[j] * (x[e] - mean[j]) * rstd[j] + b[j]);
  }
}

// operators/extra -> feat cols [0,123); zero pad cols [1147,1152)
__global__ void k_pack_feat(const float* __restrict__ ops, const float* __restrict__ ext,
                            __bf16* __restrict__ feat) {
  const size_t stride = (size_t)gridDim.x * blockDim.x;
  for (size_t e = blockIdx.x * (size_t)blockDim.x + threadIdx.x; e < (size_t)M1 * 128; e += stride) {
    size_t r = e >> 7; int j = (int)(e & 127);
    float v; int col;
    if (j < 15)       { v = ops[r * 15 + j];        col = j; }
    else if (j < 123) { v = ext[r * 108 + (j - 15)]; col = j; }
    else              { v = 0.f;                     col = j + 1024; } // 1147..1151
    feat[r * F2P + col] = (__bf16)v;
  }
}

// sample_output = relu(samples @ Ws^T + bs) * mask -> feat cols [635,1147)
__global__ void __launch_bounds__(256)
k_samples(const float* __restrict__ smp, const __bf16* __restrict__ Wsb,
          const float* __restrict__ bsv, const float* __restrict__ mask,
          __bf16* __restrict__ feat) {
  extern __shared__ char smem[];
  __bf16* AL = (__bf16*)smem;  // [32][1032] (K padded 1000->1024, +8 bank pad)
  const int tid = threadIdx.x, lane = tid & 31, wave = tid >> 5;
  const size_t r0 = (size_t)blockIdx.x * 32;
  for (int i = tid; i < 32 * 1024; i += 256) {
    int row = i >> 10, k = i & 1023;
    AL[row * 1032 + k] = (k < 1000) ? (__bf16)smp[(r0 + row) * 1000 + k] : (__bf16)0.f;
  }
  __syncthreads();
  const int mo = (lane < 16) ? 0 : 8, nn = lane & 15;
  for (int pi = 0; pi < 8; ++pi) {
    const int p = wave * 8 + pi, mb = p >> 5, cg = p & 31;
    v8f acc = vzero8();
#pragma unroll 4
    for (int ks = 0; ks < 32; ++ks) {
      v16bf a = loadA(AL + mb * 16 * 1032, 1032, ks * 32, lane);
      v16bf b = loadB(Wsb, 1024, cg * 16, ks * 32, lane);
      acc = wmma_bf16(a, b, acc);
    }
    const int col = cg * 16 + nn;
    const float bb = bsv[col];
#pragma unroll
    for (int rr = 0; rr < 8; ++rr) {
      const int row = mb * 16 + mo + rr;
      float v = fmaxf(acc[rr] + bb, 0.f) * mask[r0 + row];
      feat[(r0 + row) * F2P + 635 + col] = (__bf16)v;
    }
  }
}

// tree gather: (nh[m0]+nh[m1])/2 with index 0 = padding row
__global__ void k_gather(const int* __restrict__ mp, const float* __restrict__ hp,
                         const float* __restrict__ cp, __bf16* __restrict__ hg,
                         float* __restrict__ cgout) {
  int e = blockIdx.x * blockDim.x + threadIdx.x;
  if (e >= NN * HH) return;
  int n = e >> 9, j = e & 511;
  int m0 = mp[n * 2], m1 = mp[n * 2 + 1];
  float h0 = (m0 > 0) ? hp[(size_t)(m0 - 1) * HH + j] : 0.f;
  float h1v = (m1 > 0) ? hp[(size_t)(m1 - 1) * HH + j] : 0.f;
  float c0 = (m0 > 0) ? cp[(size_t)(m0 - 1) * HH + j] : 0.f;
  float c1v = (m1 > 0) ? cp[(size_t)(m1 - 1) * HH + j] : 0.f;
  hg[e] = (__bf16)(0.5f * (h0 + h1v));
  cgout[e] = 0.5f * (c0 + c1v);
}

// tree-level lstm cell: gates = feat @ Wih2^T + hg @ Whh2^T + b
__global__ void __launch_bounds__(256)
k_tree_lstm(const __bf16* __restrict__ featl, const __bf16* __restrict__ hg,
            const float* __restrict__ cgin, const __bf16* __restrict__ Wih,
            const __bf16* __restrict__ Whh, const float* __restrict__ bsum,
            float* __restrict__ hout, float* __restrict__ cout, int top) {
  extern __shared__ char smem[];
  float*  CG = (float*)smem;                    // [32][512]
  __bf16* FA = (__bf16*)(smem + 32 * 512 * 4);  // [32][1160]
  __bf16* HG = FA + 32 * 1160;                  // [32][520]
  const int tid = threadIdx.x, lane = tid & 31, wave = tid >> 5;
  const int r0 = blockIdx.x * 32;

  for (int i = tid; i < 32 * F2P; i += 256) {
    int row = i / F2P, k = i - row * F2P;
    FA[row * 1160 + k] = featl[(size_t)(r0 + row) * F2P + k];
  }
  for (int i = tid; i < 32 * 512; i += 256) {
    int row = i >> 9, k = i & 511;
    if (top) { HG[row * 520 + k] = (__bf16)0.f; CG[i] = 0.f; }
    else {
      HG[row * 520 + k] = hg[(size_t)(r0 + row) * 512 + k];
      CG[i] = cgin[(size_t)(r0 + row) * 512 + k];
    }
  }
  __syncthreads();
  const int mo = (lane < 16) ? 0 : 8, nn = lane & 15;
  for (int pi = 0; pi < 8; ++pi) {
    const int p = wave * 8 + pi, mb = p >> 5, cg = p & 31;
    v8f acc[4];
#pragma unroll
    for (int g = 0; g < 4; ++g) acc[g] = vzero8();
#pragma unroll 4
    for (int ks = 0; ks < F2P / 32; ++ks) {               // 36 k-steps (feat)
      v16bf a = loadA(FA + mb * 16 * 1160, 1160, ks * 32, lane);
#pragma unroll
      for (int g = 0; g < 4; ++g) {
        v16bf b = loadB(Wih, F2P, g * HH + cg * 16, ks * 32, lane);
        acc[g] = wmma_bf16(a, b, acc[g]);
      }
    }
#pragma unroll 4
    for (int ks = 0; ks < 16; ++ks) {                     // 16 k-steps (h)
      v16bf a = loadA(HG + mb * 16 * 520, 520, ks * 32, lane);
#pragma unroll
      for (int g = 0; g < 4; ++g) {
        v16bf b = loadB(Whh, HH, g * HH + cg * 16, ks * 32, lane);
        acc[g] = wmma_bf16(a, b, acc[g]);
      }
    }
    const int col = cg * 16 + nn;
    const float bi = bsum[col], bff = bsum[HH + col];
    const float bgg = bsum[2 * HH + col], boo = bsum[3 * HH + col];
#pragma unroll
    for (int rr = 0; rr < 8; ++rr) {
      const int row = mb * 16 + mo + rr;
      float zi = acc[0][rr] + bi, zf = acc[1][rr] + bff;
      float zg = acc[2][rr] + bgg, zo = acc[3][rr] + boo;
      float cv = CG[row * 512 + col];
      cv = sigmf(zf) * cv + sigmf(zi) * tanhf(zg);
      float hv = sigmf(zo) * tanhf(cv);
      hout[(size_t)(r0 + row) * 512 + col] = hv;
      cout[(size_t)(r0 + row) * 512 + col] = cv;
    }
  }
}

// generic [M,512]x[512,512]^T GEMM + bias + relu (M=1024)
__global__ void __launch_bounds__(256)
k_mlp_gemm(const __bf16* __restrict__ A, const __bf16* __restrict__ W,
           const float* __restrict__ bias, float* __restrict__ out, int M) {
  extern __shared__ char smem[];
  __bf16* AL = (__bf16*)smem;  // [32][520]
  const int tid = threadIdx.x, lane = tid & 31, wave = tid >> 5;
  const size_t r0 = (size_t)blockIdx.x * 32;
  for (int i = tid; i < 32 * 512; i += 256) {
    int row = i >> 9, k = i & 511;
    AL[row * 520 + k] = A[(r0 + row) * 512 + k];
  }
  __syncthreads();
  const int mo = (lane < 16) ? 0 : 8, nn = lane & 15;
  for (int pi = 0; pi < 8; ++pi) {
    const int p = wave * 8 + pi, mb = p >> 5, cg = p & 31;
    v8f acc = vzero8();
#pragma unroll
    for (int ks = 0; ks < 16; ++ks) {
      v16bf a = loadA(AL + mb * 16 * 520, 520, ks * 32, lane);
      v16bf b = loadB(W, HH, cg * 16, ks * 32, lane);
      acc = wmma_bf16(a, b, acc);
    }
    const int col = cg * 16 + nn;
    const float bb = bias[col];
#pragma unroll
    for (int rr = 0; rr < 8; ++rr) {
      const int row = mb * 16 + mo + rr;
      out[(r0 + row) * 512 + col] = fmaxf(acc[rr] + bb, 0.f);
    }
  }
}

// final heads: sigmoid(u @ Wo^T + bo) for both branches
__global__ void k_head(const float* __restrict__ u1, const float* __restrict__ Wo1,
                       const float* __restrict__ bo1, const float* __restrict__ u2,
                       const float* __restrict__ Wo2, const float* __restrict__ bo2,
                       float* __restrict__ out, int l) {
  int row = blockIdx.x * blockDim.x + threadIdx.x;
  if (row >= NN) return;
  float s1 = bo1[0], s2 = bo2[0];
  for (int k = 0; k < HH; ++k) {
    s1 += u1[(size_t)row * HH + k] * Wo1[k];
    s2 += u2[(size_t)row * HH + k] * Wo2[k];
  }
  out[(size_t)l * NN + row] = 1.f / (1.f + __expf(-s1));
  out[(size_t)LL * NN + (size_t)l * NN + row] = 1.f / (1.f + __expf(-s2));
}

extern "C" void kernel_launch(void* const* d_in, const int* in_sizes, int n_in,
                              void* d_out, int out_size, void* d_ws, size_t ws_size,
                              hipStream_t stream) {
  (void)in_sizes; (void)n_in; (void)out_size; (void)ws_size;
  const float* operators = (const float*)d_in[0];
  const float* extra     = (const float*)d_in[1];
  const float* cond1     = (const float*)d_in[2];
  const float* cond2     = (const float*)d_in[3];
  const float* samples   = (const float*)d_in[4];
  const float* cmask     = (const float*)d_in[5];
  const int*   mapping   = (const int*)d_in[6];
  const float* Wih1 = (const float*)d_in[7];
  const float* Whh1 = (const float*)d_in[8];
  const float* bih1 = (const float*)d_in[9];
  const float* bhh1 = (const float*)d_in[10];
  const float* Wih2 = (const float*)d_in[11];
  const float* Whh2 = (const float*)d_in[12];
  const float* bih2 = (const float*)d_in[13];
  const float* bhh2 = (const float*)d_in[14];
  const float* Wsw  = (const float*)d_in[15];
  const float* bsw  = (const float*)d_in[16];
  const float* Wcw  = (const float*)d_in[17];
  const float* bcw  = (const float*)d_in[18];
  const float* bn1g = (const float*)d_in[19];
  const float* bn1b = (const float*)d_in[20];
  const float* bn2g = (const float*)d_in[21];
  const float* bn2b = (const float*)d_in[22];
  const float* bn3g = (const float*)d_in[23];
  const float* bn3b = (const float*)d_in[24];
  const float* W2t1 = (const float*)d_in[25];
  const float* b2t1 = (const float*)d_in[26];
  const float* W2t2 = (const float*)d_in[27];
  const float* b2t2 = (const float*)d_in[28];
  const float* W3t1 = (const float*)d_in[29];
  const float* b3t1 = (const float*)d_in[30];
  const float* W3t2 = (const float*)d_in[31];
  const float* b3t2 = (const float*)d_in[32];
  const float* Wo1  = (const float*)d_in[33];
  const float* bo1  = (const float*)d_in[34];
  const float* Wo2  = (const float*)d_in[35];
  const float* bo2  = (const float*)d_in[36];

  char* base = (char*)d_ws;
  size_t off = 0;
  auto alloc = [&](size_t bytes) -> char* {
    char* pp = base + off;
    off = (off + bytes + 255) & ~(size_t)255;
    return pp;
  };

  __bf16* wih1b = (__bf16*)alloc((size_t)G4 * INW * 2);
  __bf16* whh1b = (__bf16*)alloc((size_t)G4 * HH * 2);
  __bf16* wih2b = (__bf16*)alloc((size_t)G4 * F2P * 2);
  __bf16* whh2b = (__bf16*)alloc((size_t)G4 * HH * 2);
  __bf16* wsb   = (__bf16*)alloc((size_t)HH * 1024 * 2);
  __bf16* wcb   = (__bf16*)alloc((size_t)HH * HH * 2);
  __bf16* w2t1b = (__bf16*)alloc((size_t)HH * HH * 2);
  __bf16* w2t2b = (__bf16*)alloc((size_t)HH * HH * 2);
  __bf16* w3t1b = (__bf16*)alloc((size_t)HH * HH * 2);
  __bf16* w3t2b = (__bf16*)alloc((size_t)HH * HH * 2);
  float* b1s = (float*)alloc(G4 * 4);
  float* b2s = (float*)alloc(G4 * 4);
  __bf16* h1b  = (__bf16*)alloc((size_t)B1 * HH * 2);
  float* loavg = (float*)alloc((size_t)M1 * HH * 4);
  float* mean1 = (float*)alloc(HH * 4);
  float* rstd1 = (float*)alloc(HH * 4);
  __bf16* feat = (__bf16*)alloc((size_t)M1 * F2P * 2);
  float* hidA = (float*)alloc((size_t)NN * HH * 4);
  float* hidB = (float*)alloc((size_t)NN * HH * 4);
  float* cidA = (float*)alloc((size_t)NN * HH * 4);
  float* cidB = (float*)alloc((size_t)NN * HH * 4);
  __bf16* hgb = (__bf16*)alloc((size_t)NN * HH * 2);
  float* cgf  = (float*)alloc((size_t)NN * HH * 4);
  __bf16* obf = (__bf16*)alloc((size_t)NN * HH * 2);
  float* t1f  = (float*)alloc((size_t)NN * HH * 4);
  float* t2f  = (float*)alloc((size_t)NN * HH * 4);
  __bf16* t1n = (__bf16*)alloc((size_t)NN * HH * 2);
  __bf16* t2n = (__bf16*)alloc((size_t)NN * HH * 2);
  float* u1f  = (float*)alloc((size_t)NN * HH * 4);
  float* u2f  = (float*)alloc((size_t)NN * HH * 4);
  float* mean2 = (float*)alloc(HH * 4);
  float* rstd2 = (float*)alloc(HH * 4);
  float* mean3 = (float*)alloc(HH * 4);
  float* rstd3 = (float*)alloc(HH * 4);
  float* mean4 = (float*)alloc(HH * 4);
  float* rstd4 = (float*)alloc(HH * 4);

  // ---- weight prep (bf16, K padded to /32, zero fill) ----
  k_cvt_pad<<<256, 256, 0, stream>>>(Wih1, wih1b, G4, INW, INW);
  k_cvt_pad<<<256, 256, 0, stream>>>(Whh1, whh1b, G4, HH, HH);
  k_cvt_pad<<<256, 256, 0, stream>>>(Wih2, wih2b, G4, 1147, F2P);
  k_cvt_pad<<<256, 256, 0, stream>>>(Whh2, whh2b, G4, HH, HH);
  k_cvt_pad<<<256, 256, 0, stream>>>(Wsw, wsb, HH, 1000, 1024);
  k_cvt_pad<<<256, 256, 0, stream>>>(Wcw, wcb, HH, HH, HH);
  k_cvt_pad<<<256, 256, 0, stream>>>(W2t1, w2t1b, HH, HH, HH);
  k_cvt_pad<<<256, 256, 0, stream>>>(W2t2, w2t2b, HH, HH, HH);
  k_cvt_pad<<<256, 256, 0, stream>>>(W3t1, w3t1b, HH, HH, HH);
  k_cvt_pad<<<256, 256, 0, stream>>>(W3t2, w3t2b, HH, HH, HH);
  k_bias_sum<<<8, 256, 0, stream>>>(bih1, bhh1, b1s, G4);
  k_bias_sum<<<8, 256, 0, stream>>>(bih2, bhh2, b2s, G4);

  // ---- stage 1: condition encoders ----
  const size_t lds1 = 32 * 512 * 4 + 2 * 32 * 520 * 2 + 8 * 32 * 72 * 2;  // ~165KB
  k_lstm1<<<B1 / 32, 256, lds1, stream>>>(cond1, cond2, wih1b, whh1b, b1s, h1b);

  const size_t lds2 = 2 * 32 * 520 * 2;
  k_logemm<<<M1 / 32, 256, lds2, stream>>>(h1b, wcb, bcw, loavg);
  k_colstats<<<HH, 256, 0, stream>>>(loavg, M1, HH, mean1, rstd1);
  k_bn_feat<<<640, 256, 0, stream>>>(loavg, mean1, rstd1, bn1g, bn1b, feat);
  k_pack_feat<<<256, 256, 0, stream>>>(operators, extra, feat);
  const size_t lds3 = 32 * 1032 * 2;
  k_samples<<<M1 / 32, 256, lds3, stream>>>(samples, wsb, bsw, cmask, feat);

  // ---- stage 2: tree recurrence (level l = 19..0), predict per level ----
  const size_t ldsT = 32 * 512 * 4 + 32 * 1160 * 2 + 32 * 520 * 2;  // ~169KB
  const size_t ldsG = 32 * 520 * 2;
  for (int l = LL - 1; l >= 0; --l) {
    const int s = (LL - 1) - l;
    float* hw = (s & 1) ? hidB : hidA;
    float* cw = (s & 1) ? cidB : cidA;
    float* hr = (s & 1) ? hidA : hidB;
    float* cr = (s & 1) ? cidA : cidB;
    if (l < LL - 1)
      k_gather<<<(NN * HH) / 256, 256, 0, stream>>>(mapping + (size_t)l * NN * 2,
                                                    hr, cr, hgb, cgf);
    k_tree_lstm<<<NN / 32, 256, ldsT, stream>>>(feat + (size_t)l * NN * F2P, hgb, cgf,
                                                wih2b, whh2b, b2s, hw, cw,
                                                (l == LL - 1) ? 1 : 0);
    // predict(H_l) -> cost_pred[l], card_pred[l]
    k_colstats<<<HH, 256, 0, stream>>>(hw, NN, HH, mean2, rstd2);
    k_bn_bf<<<512, 256, 0, stream>>>(hw, mean2, rstd2, bn2g, bn2b, obf, NN);
    k_mlp_gemm<<<NN / 32, 256, ldsG, stream>>>(obf, w2t1b, b2t1, t1f, NN);
    k_mlp_gemm<<<NN / 32, 256, ldsG, stream>>>(obf, w2t2b, b2t2, t2f, NN);
    k_colstats<<<HH, 256, 0, stream>>>(t1f, NN, HH, mean3, rstd3);
    k_bn_bf<<<512, 256, 0, stream>>>(t1f, mean3, rstd3, bn3g, bn3b, t1n, NN);
    k_colstats<<<HH, 256, 0, stream>>>(t2f, NN, HH, mean4, rstd4);
    k_bn_bf<<<512, 256, 0, stream>>>(t2f, mean4, rstd4, bn3g, bn3b, t2n, NN);
    k_mlp_gemm<<<NN / 32, 256, ldsG, stream>>>(t1n, w3t1b, b3t1, u1f, NN);
    k_mlp_gemm<<<NN / 32, 256, ldsG, stream>>>(t2n, w3t2b, b3t2, u2f, NN);
    k_head<<<NN / 256, 256, 0, stream>>>(u1f, Wo1, bo1, u2f, Wo2, bo2,
                                         (float*)d_out, l);
  }
}
